// Autoencoder_65120294142543
// MI455X (gfx1250) — compile-verified, matrix operand order checked by reading, not run on hardware
//
#include <hip/hip_runtime.h>
#include <hip/hip_bf16.h>

typedef __attribute__((ext_vector_type(16))) __bf16 v16bf;
typedef __attribute__((ext_vector_type(8)))  __bf16 v8bf;
typedef __attribute__((ext_vector_type(2)))  __bf16 v2bf;
typedef __attribute__((ext_vector_type(8)))  float  v8f;
typedef __attribute__((ext_vector_type(4)))  float  v4f;

#define NUM_THETAS 64
#define NUM_PTS    100
#define NGRAPH     256
#define HID        4096
#define ECT_D      4096   // 64*64
#define SCALE_F    500.0f

// ---------------------------------------------------------------------------
// ECT layer: one workgroup per graph (100 nodes). Computes normalized
// soft Euler-characteristic transform [64 bumps x 64 thetas] per graph.
// ---------------------------------------------------------------------------
template <bool OUT_BF16>
__global__ void ect_kernel(const float* __restrict__ x,    // [NGRAPH*100, 2]
                           const float* __restrict__ V,    // [2, 64]
                           const float* __restrict__ lin,  // [64]
                           float* __restrict__ outf,       // [NGRAPH, 4096] if !OUT_BF16
                           __bf16* __restrict__ outb)      // [NGRAPH, 4096] if OUT_BF16
{
    __shared__ float xs[2 * NUM_PTS];
    __shared__ float Vs[2 * NUM_THETAS];
    __shared__ float linS[NUM_THETAS];
    __shared__ float nh[NUM_PTS][NUM_THETAS];
    __shared__ float red[256];

    const int t = threadIdx.x;
    const int g = blockIdx.x;

    const float* xg = x + (size_t)g * (2 * NUM_PTS);
    if (t < 2 * NUM_PTS) xs[t] = xg[t];
    if (t < 2 * NUM_THETAS) Vs[t] = V[t];
    if (t < NUM_THETAS) linS[t] = lin[t];
    __syncthreads();

    for (int i = t; i < NUM_PTS * NUM_THETAS; i += 256) {
        const int n = i >> 6, th = i & 63;
        nh[n][th] = xs[2 * n] * Vs[th] + xs[2 * n + 1] * Vs[NUM_THETAS + th];
    }
    __syncthreads();

    float ev[16];
    float lmax = 0.0f;
#pragma unroll
    for (int j = 0; j < 16; ++j) {
        const int o = t + j * 256;
        const int bump = o >> 6, th = o & 63;
        const float L = linS[bump];
        float s = 0.0f;
        for (int n = 0; n < NUM_PTS; ++n) {
            const float z = SCALE_F * (L - nh[n][th]);
            s += 1.0f / (1.0f + __expf(-z));
        }
        ev[j] = s;
        lmax = fmaxf(lmax, s);
    }

    red[t] = lmax;
    __syncthreads();
    for (int s2 = 128; s2 > 0; s2 >>= 1) {
        if (t < s2) red[t] = fmaxf(red[t], red[t + s2]);
        __syncthreads();
    }
    const float inv = 1.0f / red[0];

    const size_t base = (size_t)g * ECT_D;
#pragma unroll
    for (int j = 0; j < 16; ++j) {
        const int o = t + j * 256;
        const float v = ev[j] * inv;
        if (OUT_BF16) outb[base + o] = (__bf16)v;
        else          outf[base + o] = v;
    }
}

// ---------------------------------------------------------------------------
// GEMM: C[M,N] = act(A[M,K](bf16) @ W[K,N](f32->bf16) + bias[N])
// Block tile 128x128, 8 waves (2x4), wave tile 64x32 = 4x2 WMMA accumulators.
// Double-buffered LDS, one barrier per k-step:
//   wait asynccnt; barrier; [async A(t+1) + W global loads(t+1)];
//   8x v_wmma on tile t; [cvt_pk + ds_store_b32 of W(t+1)]
// A tile: GLOBAL_LOAD_ASYNC_TO_LDS_B128 (ASYNCcnt, no VGPR round trip).
// W tile: f32 loads of k-row pairs -> v_cvt_pk_bf16_f32 pairs -> b32 stores
//         into K-transposed LDS so B fragments are contiguous ds_load_b128.
// ---------------------------------------------------------------------------
template <bool TANH, bool OUT_BF16, bool FULLN>
__global__ void gemm_wmma_bf16(const __bf16* __restrict__ A,   // [M,K] bf16
                               const float*  __restrict__ W,   // [K,N] f32
                               const float*  __restrict__ bias,// [N]
                               __bf16* __restrict__ outb,      // [M,N] bf16
                               float*  __restrict__ outf,      // [M,N] f32
                               int M, int N, int K)
{
    __shared__ __bf16 As[2][128][40];   // [buf][row][k] (80B row stride)
    __shared__ __bf16 Bst[2][128][40];  // [buf][col][k] (transposed W tile)

    const int tid  = threadIdx.x;
    const int lane = tid & 31;
    const int wid  = tid >> 5;
    const int waveM = wid >> 2;       // 0..1
    const int waveN = wid & 3;        // 0..3

    const int nt0 = blockIdx.x * 128;
    const int mt0 = blockIdx.y * 128;

    const int lrow = lane & 15;
    const int lk   = (lane >> 4) * 8; // lanes 16..31 hold K+8 / K+24 halves

    // ---- A-tile async copy slots (2 x 16B per thread) ----
    const int aRow0 = tid >> 2, aSeg = tid & 3;
    const int aRow1 = aRow0 + 64;
    const unsigned aLds0[2] = { (unsigned)(uintptr_t)&As[0][aRow0][aSeg * 8],
                                (unsigned)(uintptr_t)&As[1][aRow0][aSeg * 8] };
    const unsigned aLds1[2] = { (unsigned)(uintptr_t)&As[0][aRow1][aSeg * 8],
                                (unsigned)(uintptr_t)&As[1][aRow1][aSeg * 8] };
    unsigned long long gA0 =
        (unsigned long long)(uintptr_t)(A + (size_t)(mt0 + aRow0) * K + aSeg * 8);
    unsigned long long gA1 =
        (unsigned long long)(uintptr_t)(A + (size_t)(mt0 + aRow1) * K + aSeg * 8);

    // ---- W-tile staging slots: 2 slots/thread, each = k-row pair x 4 cols ----
    const int kp0 = tid >> 5,           sg0 = tid & 31;         // kp 0..7
    const int kp1 = (tid + 256) >> 5,   sg1 = (tid + 256) & 31; // kp 8..15
    const int gc0 = nt0 + sg0 * 4;
    const int gc1 = nt0 + sg1 * 4;
    const float* wp0 = W + (size_t)(2 * kp0) * N + gc0;
    const float* wp1 = W + (size_t)(2 * kp1) * N + gc1;
    const size_t wAdv = (size_t)32 * N;

    v8f acc[4][2];
#pragma unroll
    for (int i = 0; i < 4; ++i)
#pragma unroll
        for (int j = 0; j < 2; ++j)
            acc[i][j] = (v8f){0.f, 0.f, 0.f, 0.f, 0.f, 0.f, 0.f, 0.f};

    auto stageA = [&](int b) {
        asm volatile("global_load_async_to_lds_b128 %0, %1, off"
                     :: "v"(aLds0[b]), "v"(gA0) : "memory");
        asm volatile("global_load_async_to_lds_b128 %0, %1, off"
                     :: "v"(aLds1[b]), "v"(gA1) : "memory");
        gA0 += 64;  // 32 bf16
        gA1 += 64;
    };
    auto loadW = [&](v4f& a0, v4f& a1, v4f& b0, v4f& b1) {
        if (FULLN) {
            a0 = *reinterpret_cast<const v4f*>(wp0);
            a1 = *reinterpret_cast<const v4f*>(wp0 + N);
            b0 = *reinterpret_cast<const v4f*>(wp1);
            b1 = *reinterpret_cast<const v4f*>(wp1 + N);
        } else {  // N assumed multiple of 4
            const v4f z = (v4f){0.f, 0.f, 0.f, 0.f};
            const bool ok0 = (gc0 + 4 <= N), ok1 = (gc1 + 4 <= N);
            a0 = ok0 ? *reinterpret_cast<const v4f*>(wp0)     : z;
            a1 = ok0 ? *reinterpret_cast<const v4f*>(wp0 + N) : z;
            b0 = ok1 ? *reinterpret_cast<const v4f*>(wp1)     : z;
            b1 = ok1 ? *reinterpret_cast<const v4f*>(wp1 + N) : z;
        }
        wp0 += wAdv;
        wp1 += wAdv;
    };
    auto storeW = [&](int b, v4f a0, v4f a1, v4f b0, v4f b1) {
#pragma unroll
        for (int c = 0; c < 4; ++c) {
            const v2bf p0 = { (__bf16)a0[c], (__bf16)a1[c] };  // (k, k+1) pair
            *reinterpret_cast<v2bf*>(&Bst[b][sg0 * 4 + c][2 * kp0]) = p0;
            const v2bf p1 = { (__bf16)b0[c], (__bf16)b1[c] };
            *reinterpret_cast<v2bf*>(&Bst[b][sg1 * 4 + c][2 * kp1]) = p1;
        }
    };
    auto compute = [&](int b) {
#pragma unroll
        for (int i = 0; i < 4; ++i) {
            const int arow = waveM * 64 + i * 16 + lrow;
            const v8bf alo = *reinterpret_cast<const v8bf*>(&As[b][arow][lk]);
            const v8bf ahi = *reinterpret_cast<const v8bf*>(&As[b][arow][lk + 16]);
            const v16bf a = __builtin_shufflevector(alo, ahi,
                0, 1, 2, 3, 4, 5, 6, 7, 8, 9, 10, 11, 12, 13, 14, 15);
#pragma unroll
            for (int j = 0; j < 2; ++j) {
                const int bcol = waveN * 32 + j * 16 + lrow;
                const v8bf blo = *reinterpret_cast<const v8bf*>(&Bst[b][bcol][lk]);
                const v8bf bhi = *reinterpret_cast<const v8bf*>(&Bst[b][bcol][lk + 16]);
                const v16bf bb = __builtin_shufflevector(blo, bhi,
                    0, 1, 2, 3, 4, 5, 6, 7, 8, 9, 10, 11, 12, 13, 14, 15);
                acc[i][j] = __builtin_amdgcn_wmma_f32_16x16x32_bf16(
                    false, a, false, bb, (short)0, acc[i][j], false, false);
            }
        }
    };

    // ---- prologue: stage tile 0 into buffer 0
    {
        v4f a0, a1, b0, b1;
        stageA(0);
        loadW(a0, a1, b0, b1);
        storeW(0, a0, a1, b0, b1);
    }

    int buf = 0;
    for (int k0 = 0; k0 < K; k0 += 32, buf ^= 1) {
        asm volatile("s_wait_asynccnt 0x0" ::: "memory");
        __syncthreads();

        const bool hasNext = (k0 + 32) < K;
        v4f a0, a1, b0, b1;
        if (hasNext) {
            stageA(buf ^ 1);          // async -> other buffer
            loadW(a0, a1, b0, b1);    // global f32 loads issue now...
        }
        compute(buf);                 // ...and retire under the 8 WMMAs
        if (hasNext) {
            storeW(buf ^ 1, a0, a1, b0, b1);
        }
    }

    // ---- epilogue: C/D layout -> lanes 0..15: M=r, N=lane; lanes 16..31: M=r+8
    const int crow = (lane >> 4) * 8;
    const int ccol = lane & 15;
#pragma unroll
    for (int i = 0; i < 4; ++i) {
#pragma unroll
        for (int j = 0; j < 2; ++j) {
#pragma unroll
            for (int r = 0; r < 8; ++r) {
                const int grow = mt0 + waveM * 64 + i * 16 + crow + r;
                const int gcol = nt0 + waveN * 32 + j * 16 + ccol;
                if (FULLN || gcol < N) {
                    float v = acc[i][j][r] + bias[gcol];
                    if (TANH) v = tanhf(v);
                    const size_t o = (size_t)grow * N + gcol;
                    if (OUT_BF16) outb[o] = (__bf16)v;
                    else          outf[o] = v;
                }
            }
        }
    }
}

// ---------------------------------------------------------------------------
extern "C" void kernel_launch(void* const* d_in, const int* in_sizes, int n_in,
                              void* d_out, int out_size, void* d_ws, size_t ws_size,
                              hipStream_t stream) {
    const float* x   = (const float*)d_in[0];
    // d_in[1] = batch_idx (int32) -- implicit in the blocked layout, unused
    const float* V   = (const float*)d_in[2];
    const float* lin = (const float*)d_in[3];
    const float* W1  = (const float*)d_in[4];
    const float* b1  = (const float*)d_in[5];
    const float* W2  = (const float*)d_in[6];
    const float* b2  = (const float*)d_in[7];
    const float* W3  = (const float*)d_in[8];
    const float* b3  = (const float*)d_in[9];

    // workspace: three bf16 activation buffers [256 x 4096] each (6 MB total)
    __bf16* ect_bf = (__bf16*)d_ws;
    __bf16* h1     = ect_bf + (size_t)NGRAPH * ECT_D;
    __bf16* h2     = h1     + (size_t)NGRAPH * HID;

    // output: decoded [256*4096] f32, then pts [25600*2] f32
    float* decoded = (float*)d_out;
    float* pts     = decoded + (size_t)NGRAPH * ECT_D;

    // 1) ECT on input points -> bf16 activations
    ect_kernel<true><<<NGRAPH, 256, 0, stream>>>(x, V, lin, nullptr, ect_bf);

    // 2) h1 = tanh(ect @ W1 + b1)
    gemm_wmma_bf16<true, true, true><<<dim3(HID / 128, 2), 256, 0, stream>>>(
        ect_bf, W1, b1, h1, nullptr, NGRAPH, HID, ECT_D);

    // 3) h2 = tanh(h1 @ W2 + b2)
    gemm_wmma_bf16<true, true, true><<<dim3(HID / 128, 2), 256, 0, stream>>>(
        h1, W2, b2, h2, nullptr, NGRAPH, HID, HID);

    // 4) pts = h2 @ W3 + b3  (N = 200, written straight into d_out tail)
    gemm_wmma_bf16<false, false, false><<<dim3(2, 2), 256, 0, stream>>>(
        h2, W3, b3, nullptr, pts, NGRAPH, 2 * NUM_PTS, HID);

    // 5) ECT on decoded points -> f32 output
    ect_kernel<false><<<NGRAPH, 256, 0, stream>>>(pts, V, lin, decoded, nullptr);
}
